// CharLSTM_20418274525283
// MI455X (gfx1250) — matrix operand order repaired and checked by reading, not compiled, not run on hardware
//
#include <hip/hip_runtime.h>
#include <stdint.h>

// Problem sizes (match reference)
static constexpr int BB = 16;     // batch
static constexpr int TT = 256;    // timesteps
static constexpr int HH = 1024;   // hidden
static constexpr int VV = 32000;  // vocab
static constexpr int K2 = 2048;   // fused K = [x | h]
static constexpr int N4 = 4096;   // fused gate cols (i|f|o|c)

// ---------------- types ----------------
typedef __bf16   bf16;
typedef bf16     v16bf __attribute__((ext_vector_type(16)));
typedef float    v8f   __attribute__((ext_vector_type(8)));
typedef uint32_t u32x4 __attribute__((ext_vector_type(4)));
typedef uint32_t u32x8 __attribute__((ext_vector_type(8)));

union Frag { u32x4 q[2]; v16bf v; };

static __device__ __forceinline__ uint16_t f2bf(float f) {
  uint32_t x = __builtin_bit_cast(uint32_t, f);
  x += 0x7FFFu + ((x >> 16) & 1u);          // round-to-nearest-even
  return (uint16_t)(x >> 16);
}

// A fragment (16x32 bf16, MxK). rowbase -> &A[m][k0]. Contiguous k layout used here (matches B, not AMD's table order):
// lanes 0-15:  VGPR0-7 = K[0..15]
// lanes 16-31: VGPR0-7 = K[16..31]
static __device__ __forceinline__ v16bf load_a_frag(const uint16_t* rowbase, int sel) {
  Frag f;
  f.q[0] = *(const u32x4*)(rowbase + sel * 16);
  f.q[1] = *(const u32x4*)(rowbase + sel * 16 + 8);
  return f.v;
}

// B fragment (32x16 bf16, KxN), weights stored transposed [N][K] so each
// lane reads 32 contiguous bytes: p -> &W[n][k0 + sel*16]
static __device__ __forceinline__ v16bf load_b_frag(const uint16_t* p) {
  Frag f;
  f.q[0] = *(const u32x4*)(p);
  f.q[1] = *(const u32x4*)(p + 8);
  return f.v;
}

static __device__ __forceinline__ v8f wmma_bf16(v16bf a, v16bf b, v8f c) {
  return __builtin_amdgcn_wmma_f32_16x16x32_bf16(false, a, false, b, (short)0, c,
                                                 false, false);
}

static __device__ __forceinline__ float sigm(float x) {
  return 1.0f / (1.0f + __expf(-x));
}

// ---- TDM: issue a 2D tile load (64 rows x 256 bf16) of Wq[N=32000][K=1024]
// into LDS. Descriptor per CDNA5 ISA ch.8 (D# groups 0..3). Wave-uniform.
static __device__ __forceinline__ void tdm_issue_b(const uint16_t* gsrc,
                                                   uint32_t ldsAddr) {
  uint64_t ga = (uint64_t)(uintptr_t)gsrc;
  u32x4 g0 = {
      1u,                                                   // count=1, user mode
      ldsAddr,                                              // lds_addr [63:32]
      (uint32_t)ga,                                         // global_addr lo
      (uint32_t)((ga >> 32) & 0x01FFFFFFu) | (2u << 30)     // ga hi | type=2
  };
  u32x8 g1 = {
      1u << 16,                      // data_size=1 (2 bytes), no multicast
      (1024u & 0xFFFFu) << 16,       // tensor_dim0[15:0]  @ bits 63:48
      (32000u & 0xFFFFu) << 16,      // dim0 hi=0 | tensor_dim1[15:0] @ 95:80
      (256u << 16),                  // dim1 hi=0 | tile_dim0=256 @ 127:112
      64u,                           // tile_dim1=64, tile_dim2=0
      1024u,                         // tensor_dim0_stride lo (elements)
      0u,                            // stride hi | dim1_stride lo
      0u
  };
  u32x4 g2 = {0u, 0u, 0u, 0u};
  u32x4 g3 = {0u, 0u, 0u, 0u};
  asm volatile("tensor_load_to_lds %0, %1, %2, %3"
               :: "s"(g0), "s"(g1), "s"(g2), "s"(g3) : "memory");
}

// ---------------- kernel 1: embedding gather -> bf16 X[t][b][h] ----------------
__global__ void k_embed(const long long* __restrict__ tok,
                        const float* __restrict__ emb,
                        uint16_t* __restrict__ X) {
  int tb = blockIdx.x;            // 0..T*B-1
  int t = tb >> 4, b = tb & 15;   // BB == 16
  long long id = tok[(size_t)b * TT + t];
  const float* src = emb + (size_t)id * HH;
  uint16_t* dst = X + ((size_t)t * BB + b) * HH;
  for (int i = threadIdx.x; i < HH; i += blockDim.x) dst[i] = f2bf(src[i]);
}

// ---------------- kernel 2: tiled fp32->bf16 transpose ----------------
// dst[(nofs+n)*dstld + kofs+k] = src[k*srcld + n]
__global__ void k_transpose_bf16(const float* __restrict__ src, int srcld,
                                 uint16_t* __restrict__ dst, int dstld,
                                 int nofs, int kofs) {
  __shared__ float tile[32][33];
  int n0 = blockIdx.x * 32, k0 = blockIdx.y * 32;
  int tx = threadIdx.x, ty = threadIdx.y;   // block (32, 8)
#pragma unroll
  for (int i = 0; i < 32; i += 8)
    tile[ty + i][tx] = src[(size_t)(k0 + ty + i) * srcld + (n0 + tx)];
  __syncthreads();
#pragma unroll
  for (int i = 0; i < 32; i += 8)
    dst[(size_t)(nofs + n0 + ty + i) * dstld + (kofs + k0 + tx)] =
        f2bf(tile[tx][ty + i]);
}

// ---------------- kernel 3: persistent LSTM recurrence ----------------
// grid = 64 WGs (hidden tiles of 16), block = 128 (4 waves = 4 gates).
// Each WG's fused-weight slice (4 gates x 16 cols x 2048 K bf16 = 262144 B)
// is async-copied once into the 320 KB WGP LDS and stays resident 256 steps.
__global__ void __launch_bounds__(128)
k_lstm(const uint16_t* __restrict__ X,      // [T][B][H] bf16
       const uint16_t* __restrict__ Wg,     // [4096][2048] bf16 (transposed)
       uint16_t* __restrict__ Hall,         // [T+1][B][H] bf16, row 0 zeroed
       const float* __restrict__ bI, const float* __restrict__ bF,
       const float* __restrict__ bO, const float* __restrict__ bC,
       unsigned* __restrict__ counter,
       float* __restrict__ outH, float* __restrict__ outC) {
  extern __shared__ char smem[];
  uint16_t* ldsW = (uint16_t*)smem;                   // 262144 B weights
  float* gbuf = (float*)(smem + 262144);              // 4 x 256 fp32 gate tiles
  float* cst  = (float*)(smem + 262144 + 4096);       // 256 fp32 cell state

  const int blk  = blockIdx.x;          // hidden tile 0..63
  const int wave = threadIdx.x >> 5;    // gate 0..3 (i,f,o,c)
  const int lane = threadIdx.x & 31;
  const int sel  = lane >> 4;
  const int mrow = lane & 15;

  // ---- async preload of this wave's 16 weight rows (64 KB) into LDS ----
  {
    uint32_t ldsBase = (uint32_t)(uintptr_t)ldsW + (uint32_t)wave * 65536u;
    const uint16_t* gsrc = Wg + ((size_t)wave * HH + (size_t)blk * 16) * K2;
#pragma unroll 4
    for (int i = 0; i < 128; ++i) {
      uint32_t off = (uint32_t)i * 512u + (uint32_t)lane * 16u;
      uint32_t ldsAddr = ldsBase + off;
      uint64_t ga = (uint64_t)(uintptr_t)((const char*)gsrc + off);
      asm volatile("global_load_async_to_lds_b128 %0, %1, off"
                   :: "v"(ldsAddr), "v"(ga) : "memory");
    }
    asm volatile("s_wait_asynccnt 0x0" ::: "memory");
  }

  // cell state init + per-thread bias registers
  for (int e = threadIdx.x; e < 256; e += 128) cst[e] = 0.0f;
  const int colT = blk * 16 + (threadIdx.x & 15);
  const float rbI = bI[colT], rbF = bF[colT], rbO = bO[colT], rbC = bC[colT];
  __syncthreads();

  const uint16_t* ldsRow = ldsW + (size_t)wave * 32768 + (size_t)mrow * K2;

  for (int t = 0; t < TT; ++t) {
    const uint16_t* axr = X    + (size_t)t * (BB * HH) + mrow * HH; // x_t row
    const uint16_t* ahr = Hall + (size_t)t * (BB * HH) + mrow * HH; // H_{t-1}
    v8f acc = {};
#pragma unroll 4
    for (int kb = 0; kb < 32; ++kb) {           // x half: K = 0..1023
      const int k0 = kb * 32;
      acc = wmma_bf16(load_a_frag(axr + k0, sel),
                      load_b_frag(ldsRow + k0 + sel * 16), acc);
    }
#pragma unroll 4
    for (int kb = 0; kb < 32; ++kb) {           // h half: K = 1024..2047
      const int k0 = kb * 32;
      acc = wmma_bf16(load_a_frag(ahr + k0, sel),
                      load_b_frag(ldsRow + HH + k0 + sel * 16), acc);
    }
    // scatter accumulator tile (M=batch, N=hidden col) into LDS
#pragma unroll
    for (int r = 0; r < 8; ++r) {
      int m = sel ? (8 + r) : r;
      gbuf[wave * 256 + m * 16 + mrow] = acc[r];
    }
    __syncthreads();

    // elementwise gate math: 256 elements, 2 per thread
    uint16_t* Hnext = Hall + (size_t)(t + 1) * (BB * HH);
#pragma unroll
    for (int h = 0; h < 2; ++h) {
      int ee = threadIdx.x + h * 128;
      float gi = gbuf[ee]       + rbI;
      float gf = gbuf[256 + ee] + rbF;
      float go = gbuf[512 + ee] + rbO;
      float gc = gbuf[768 + ee] + rbC;
      float I = sigm(gi), F = sigm(gf), O = sigm(go);
      float Ct = tanhf(gc);
      float Cn = F * cst[ee] + I * Ct;
      cst[ee] = Cn;
      float Hn = O * tanhf(Cn);
      int m = ee >> 4, col = blk * 16 + (ee & 15);
      Hnext[m * HH + col] = f2bf(Hn);
      if (t == TT - 1) {
        outH[m * HH + col] = Hn;
        outC[m * HH + col] = Cn;
      }
    }

    // -------- device-wide phase barrier (64 co-resident WGs) --------
    __syncthreads();
    if (threadIdx.x == 0) {
      __threadfence();
      __hip_atomic_fetch_add(counter, 1u, __ATOMIC_RELEASE,
                             __HIP_MEMORY_SCOPE_AGENT);
      const unsigned tgt = (unsigned)(t + 1) * gridDim.x;
      while (__hip_atomic_load(counter, __ATOMIC_ACQUIRE,
                               __HIP_MEMORY_SCOPE_AGENT) < tgt)
        __builtin_amdgcn_s_sleep(1);
    }
    __syncthreads();
  }
}

// ---------------- kernel 4: output projection Y = H_all @ W_hq + b_q ----------------
// M = T*B = 4096, N = 32000, K = 1024. Block = 8 waves sharing one 64-col B
// panel staged in LDS by the Tensor Data Mover (double-buffered 2x32KB);
// each wave owns a 16x64 output tile -> block tile 128M x 64N. This cuts
// W_hq L2 traffic 8x vs per-wave streaming (~2 GB total).
__global__ void __launch_bounds__(256)
k_proj(const uint16_t* __restrict__ Hall1,   // [T*B][H] bf16 (skips zero row)
       const uint16_t* __restrict__ Wq,      // [V][H] bf16 (transposed)
       const float* __restrict__ bq,
       float* __restrict__ Y) {              // [B][T][V] fp32
  extern __shared__ char smem[];
  uint16_t* buf = (uint16_t*)smem;           // 2 x (64 rows x 256 K) bf16
  const int wave = threadIdx.x >> 5, lane = threadIdx.x & 31;
  const int sel = lane >> 4, nr = lane & 15;
  const int n0 = blockIdx.x * 64;
  const int m0 = blockIdx.y * 128 + wave * 16;

  const uint32_t ldsBase = (uint32_t)(uintptr_t)buf;
  const uint16_t* gB = Wq + (size_t)n0 * HH;       // chunk c starts at +c*256

  if (wave == 0) tdm_issue_b(gB, ldsBase);         // prefetch chunk 0

  v8f acc0 = {}, acc1 = {}, acc2 = {}, acc3 = {};
  const uint16_t* arow = Hall1 + (size_t)(m0 + nr) * HH;

  for (int c = 0; c < 4; ++c) {                    // K chunks of 256
    if (wave == 0) {
      if (c + 1 < 4) {
        tdm_issue_b(gB + (c + 1) * 256, ldsBase + (uint32_t)(((c + 1) & 1) * 32768));
        __builtin_amdgcn_s_wait_tensorcnt(1);      // chunk c has landed
      } else {
        __builtin_amdgcn_s_wait_tensorcnt(0);      // last chunk has landed
      }
    }
    __syncthreads();                               // chunk c visible to all
    const uint16_t* bchunk = buf + ((c & 1) ? 16384 : 0);
#pragma unroll
    for (int kk = 0; kk < 8; ++kk) {
      const int kl = kk * 32;
      v16bf a = load_a_frag(arow + c * 256 + kl, sel);
      const int ko = kl + sel * 16;
      acc0 = wmma_bf16(a, load_b_frag(bchunk + (size_t)(nr)      * 256 + ko), acc0);
      acc1 = wmma_bf16(a, load_b_frag(bchunk + (size_t)(16 + nr) * 256 + ko), acc1);
      acc2 = wmma_bf16(a, load_b_frag(bchunk + (size_t)(32 + nr) * 256 + ko), acc2);
      acc3 = wmma_bf16(a, load_b_frag(bchunk + (size_t)(48 + nr) * 256 + ko), acc3);
    }
    __syncthreads();                               // done reading buf[c&1]
  }

  const float q0 = bq[n0 + nr], q1 = bq[n0 + 16 + nr];
  const float q2 = bq[n0 + 32 + nr], q3 = bq[n0 + 48 + nr];
#pragma unroll
  for (int r = 0; r < 8; ++r) {
    int m = m0 + (sel ? 8 + r : r);
    int t = m >> 4, b = m & 15;              // row m = t*16 + b
    size_t base = ((size_t)b * TT + t) * VV;
    Y[base + n0 + nr]      = acc0[r] + q0;
    Y[base + n0 + 16 + nr] = acc1[r] + q1;
    Y[base + n0 + 32 + nr] = acc2[r] + q2;
    Y[base + n0 + 48 + nr] = acc3[r] + q3;
  }
}

// ---------------- workspace layout ----------------
static constexpr size_t OFF_CNT  = 0;                                   // 256 B
static constexpr size_t OFF_HALL = 256;
static constexpr size_t SZ_HALL  = (size_t)(TT + 1) * BB * HH * 2;      // 8,421,376
static constexpr size_t OFF_X    = OFF_HALL + SZ_HALL;
static constexpr size_t SZ_X     = (size_t)TT * BB * HH * 2;            // 8,388,608
static constexpr size_t OFF_WG   = OFF_X + SZ_X;
static constexpr size_t SZ_WG    = (size_t)N4 * K2 * 2;                 // 16,777,216
static constexpr size_t OFF_WQ   = OFF_WG + SZ_WG;
static constexpr size_t SZ_WQ    = (size_t)VV * HH * 2;                 // 65,536,000

extern "C" void kernel_launch(void* const* d_in, const int* in_sizes, int n_in,
                              void* d_out, int out_size, void* d_ws, size_t ws_size,
                              hipStream_t stream) {
  (void)in_sizes; (void)n_in; (void)out_size; (void)ws_size;

  const long long* tok = (const long long*)d_in[0];
  const float* emb = (const float*)d_in[1];
  const float* Wxi = (const float*)d_in[2];
  const float* Whi = (const float*)d_in[3];
  const float* bI  = (const float*)d_in[4];
  const float* Wxf = (const float*)d_in[5];
  const float* Whf = (const float*)d_in[6];
  const float* bF  = (const float*)d_in[7];
  const float* Wxo = (const float*)d_in[8];
  const float* Who = (const float*)d_in[9];
  const float* bO  = (const float*)d_in[10];
  const float* Wxc = (const float*)d_in[11];
  const float* Whc = (const float*)d_in[12];
  const float* bC  = (const float*)d_in[13];
  const float* Whq = (const float*)d_in[14];
  const float* bq  = (const float*)d_in[15];

  char* ws = (char*)d_ws;
  unsigned* counter = (unsigned*)(ws + OFF_CNT);
  uint16_t* Hall = (uint16_t*)(ws + OFF_HALL);
  uint16_t* X    = (uint16_t*)(ws + OFF_X);
  uint16_t* Wg   = (uint16_t*)(ws + OFF_WG);
  uint16_t* WqT  = (uint16_t*)(ws + OFF_WQ);

  float* Y    = (float*)d_out;
  float* outH = Y + (size_t)BB * TT * VV;
  float* outC = outH + (size_t)BB * HH;

  // zero barrier counter + H_{-1} row (graph-capture-safe)
  (void)hipMemsetAsync(ws, 0, 256 + (size_t)BB * HH * 2, stream);

  // embedding gather -> bf16
  k_embed<<<dim3(TT * BB), dim3(256), 0, stream>>>(tok, emb, X);

  // fuse+transpose gate weights into Wg[N=4096][K=2048] (i|f|o|c, x then h)
  const float* srcs[8] = {Wxi, Wxf, Wxo, Wxc, Whi, Whf, Who, Whc};
  for (int z = 0; z < 8; ++z) {
    int g = z & 3, xh = z >> 2;
    k_transpose_bf16<<<dim3(HH / 32, HH / 32), dim3(32, 8), 0, stream>>>(
        srcs[z], HH, Wg, K2, g * HH, xh * HH);
  }
  // transpose W_hq [1024][32000] -> WqT [32000][1024] bf16
  k_transpose_bf16<<<dim3(VV / 32, HH / 32), dim3(32, 8), 0, stream>>>(
      Whq, VV, WqT, HH, 0, 0);

  // persistent recurrence: 64 WGs x 4 waves, 262 KB weights + 5 KB scratch in LDS
  size_t smem = 262144 + 4096 + 1024;
  k_lstm<<<dim3(64), dim3(128), smem, stream>>>(X, Wg, Hall, bI, bF, bO, bC,
                                                counter, outH, outC);

  // projection GEMM: 268 GFLOP of WMMA, TDM-staged B panels (64 KB LDS)
  k_proj<<<dim3(VV / 64, (TT * BB) / 128), dim3(256), 65536, stream>>>(
      Hall + (size_t)BB * HH, WqT, bq, Y);
}
